// ResNet_33638183862326
// MI455X (gfx1250) — matrix operand order repaired and checked by reading, not compiled, not run on hardware
//
#include <hip/hip_runtime.h>
#include <stdint.h>

// ---------------------------------------------------------------------------
// CDNA5 (gfx1250) vector types
// ---------------------------------------------------------------------------
typedef __attribute__((ext_vector_type(16))) __bf16        v16bf;
typedef __attribute__((ext_vector_type(8)))  float         v8f;
typedef __attribute__((ext_vector_type(4)))  unsigned int  v4u;

__device__ __forceinline__ unsigned short f2bf(float f) {
    // round-to-nearest-even f32 -> bf16
    unsigned int u = __float_as_uint(f);
    unsigned int r = u + 0x7FFFu + ((u >> 16) & 1u);
    return (unsigned short)(r >> 16);
}
__device__ __forceinline__ float bf2f(unsigned short h) {
    return __uint_as_float((unsigned int)h << 16);
}

// ---------------------------------------------------------------------------
// misc small kernels
// ---------------------------------------------------------------------------
__global__ void k_zero(float* __restrict__ p, int n) {
    int i = blockIdx.x * blockDim.x + threadIdx.x;
    if (i < n) p[i] = 0.0f;
}

// initial 5^3 submanifold conv, Cin=1 -> Cout=32 (K-dim too thin for WMMA)
// writes bf16 activations
__global__ __launch_bounds__(128)
void k_conv_init(const float* __restrict__ feats, const int* __restrict__ nbr,
                 const float* __restrict__ w /*[125][32]*/,
                 unsigned short* __restrict__ y, int N) {
    int site = blockIdx.x * 4 + (threadIdx.x >> 5);
    int co   = threadIdx.x & 31;
    if (site >= N) return;
    float acc = 0.0f;
    const int* nb = nbr + (size_t)site * 125;
    for (int k = 0; k < 125; ++k) {
        int idx = nb[k];
        float f = (idx >= 0) ? feats[idx] : 0.0f;
        acc = fmaf(f, w[k * 32 + co], acc);
    }
    y[(size_t)site * 32 + co] = f2bf(acc);
}

// Repack conv weight W[K][Ci][Co] (f32, row-major) into per-lane bf16 B-fragments:
// dst[(((k*CC + cc)*NT + nt)*32 + lane)*16 + j] =
//     bf16( W[k][cc*32 + 8*(lane>>4) + (j&7) + 16*(j>>3)][nt*16 + (lane&15)] )
// (WMMA B is 32x16: lanes 0-15 hold K 0-7,16-23; lanes 16-31 hold K 8-15,24-31)
__global__ void k_repack_w(const float* __restrict__ w, unsigned short* __restrict__ dst,
                           int K, int Ci, int Co) {
    int total = K * Ci * Co;
    int i = blockIdx.x * blockDim.x + threadIdx.x;
    if (i >= total) return;
    int NT = Co >> 4, CC = Ci >> 5;
    int j    = i & 15;
    int lane = (i >> 4) & 31;
    int rest = i >> 9;
    int nt = rest % NT; rest /= NT;
    int cc = rest % CC;
    int k  = rest / CC;
    int kk = cc * 32 + ((lane >> 4) << 3) + (j & 7) + ((j >> 3) << 4);  // cin index
    int n  = nt * 16 + (lane & 15);             // cout index
    dst[i] = f2bf(w[((size_t)k * Ci + kk) * Co + n]);
}

// ---------------------------------------------------------------------------
// Sparse gather-conv, implicit GEMM over rulebook taps, bf16 WMMA f32-accum.
// Block = 128 threads (4 waves) = 64 output sites. Wave w owns sites
// [site0+16w, site0+16w+16) (its own A fragments) and loops over ALL cout
// tiles, so every wave issues WMMAs regardless of Cout.
// Gather staging uses CDNA5 async global->LDS copies (GLOBAL_LOAD_ASYNC_TO_LDS
// _B128, tracked on ASYNCcnt), double-buffered so tap k+1 streams in while tap
// k is consumed by WMMAs. Inactive rulebook entries gather from a zero row so
// EXEC stays uniform. x is bf16-resident (halves gather traffic; all activa-
// tions are L2-resident on MI455X's 192MB L2, HBM stays idle).
// ---------------------------------------------------------------------------
template<int CIN, int COUT, int K>
__global__ __launch_bounds__(128)
void k_conv_wmma(const unsigned short* __restrict__ x, const int* __restrict__ nbr,
                 const unsigned short* __restrict__ wt,
                 const unsigned short* __restrict__ zrow,
                 float* __restrict__ y, int Nout) {
    constexpr int STRIDE = CIN + 8;        // ushorts; multiple of 8 -> 16B rows
    constexpr int CC = CIN / 32;           // 32-wide cin chunks
    constexpr int NT = COUT / 16;          // 16-wide cout tiles
    __shared__ unsigned short lds[2][64 * STRIDE];

    const int tid   = threadIdx.x;
    const int wave  = tid >> 5;
    const int lane  = tid & 31;
    const int site0 = blockIdx.x * 64;

    v8f acc[NT];
    #pragma unroll
    for (int t = 0; t < NT; ++t)
        #pragma unroll
        for (int r = 0; r < 8; ++r) acc[t][r] = 0.0f;

    // gather staging: 2 threads per row, each covers CIN/2 contiguous bf16
    // channels = CIN bytes = CIN/16 async b128 chunks
    const int srow  = tid >> 1;
    const int scol0 = (tid & 1) * (CIN / 2);

    auto stage = [&](int k, int buf) {
        int gsite = site0 + srow;
        int idx = (gsite < Nout) ? nbr[(size_t)gsite * K + k] : -1;
        const unsigned short* src =
            (idx >= 0) ? (x + (size_t)idx * CIN + scol0) : zrow;
        // low 32 bits of a generic pointer to __shared__ == LDS byte address
        unsigned int dstoff =
            (unsigned int)(uintptr_t)(&lds[buf][srow * STRIDE + scol0]);
        #pragma unroll
        for (int c = 0; c < CIN / 16; ++c) {
            // INST_OFFSET is added to BOTH the LDS and global addresses
            asm volatile("global_load_async_to_lds_b128 %0, %1, off offset:%2"
                         :: "v"(dstoff), "v"(src), "i"(c * 16)
                         : "memory");
        }
    };

    // A fragment addressing (ISA 16-bit A 16x32 layout):
    // lane L, vgpr v, half h -> k = (v>=4?16:0) + (L>=16?8:0) + 2*(v&3) + h
    // => per lane: halves [khi, khi+8) and [khi+16, khi+24) : two b128 LDS loads
    const int arow = wave * 16 + (lane & 15);
    const int khi  = (lane >> 4) << 3;

    stage(0, 0);

    #pragma unroll 1
    for (int k = 0; k < K; ++k) {
        // our async writes must land before the workgroup barrier publishes them
        asm volatile("s_wait_asynccnt 0x0" ::: "memory");
        __syncthreads();                       // staged tap k visible to all
        if (k + 1 < K) stage(k + 1, (k + 1) & 1);
        const unsigned short* abuf = &lds[k & 1][arow * STRIDE + khi];
        #pragma unroll
        for (int cc = 0; cc < CC; ++cc) {
            union { v16bf v; v4u q[2]; } A;
            A.q[0] = *reinterpret_cast<const v4u*>(abuf + cc * 32);
            A.q[1] = *reinterpret_cast<const v4u*>(abuf + cc * 32 + 16);
            #pragma unroll
            for (int nt = 0; nt < NT; ++nt) {
                const v16bf B = *reinterpret_cast<const v16bf*>(
                    wt + (((size_t)(k * CC + cc) * NT + nt) * 32 + lane) * 16);
                acc[nt] = __builtin_amdgcn_wmma_f32_16x16x32_bf16(
                    false, A.v, false, B, (short)0, acc[nt], false, false);
            }
        }
    }

    // D store (f32, for exact BN stats): vgpr r -> M = r + 8*(lane>=16), N = lane&15
    const int m0 = wave * 16 + ((lane >> 4) << 3);
    const int n  = lane & 15;
    #pragma unroll
    for (int nt = 0; nt < NT; ++nt)
        #pragma unroll
        for (int r = 0; r < 8; ++r) {
            int site = site0 + m0 + r;
            if (site < Nout) y[(size_t)site * COUT + nt * 16 + n] = acc[nt][r];
        }
}

// ---------------------------------------------------------------------------
// BatchNorm: per-channel sum / sumsq reduction (LDS atomics -> global atomics)
// stats[0..C-1] = sum, stats[128..128+C-1] = sumsq  (must be pre-zeroed)
// ---------------------------------------------------------------------------
__global__ __launch_bounds__(256)
void k_bn_stats(const float* __restrict__ x, float* __restrict__ stats, int N, int C) {
    __shared__ float s_sum[128], s_sq[128];
    for (int c = threadIdx.x; c < C; c += blockDim.x) { s_sum[c] = 0.0f; s_sq[c] = 0.0f; }
    __syncthreads();
    int total = N * C;
    for (int i = blockIdx.x * blockDim.x + threadIdx.x; i < total;
         i += gridDim.x * blockDim.x) {
        float v = x[i];
        int c = i % C;
        atomicAdd(&s_sum[c], v);
        atomicAdd(&s_sq[c], v * v);
    }
    __syncthreads();
    for (int c = threadIdx.x; c < C; c += blockDim.x) {
        atomicAdd(&stats[c], s_sum[c]);
        atomicAdd(&stats[128 + c], s_sq[c]);
    }
}

// y(bf16) = bn(x_f32)*g + b [+ res(bf16)] [relu]; biased var, eps=1e-4
__global__ void k_bn_apply(const float* __restrict__ x, const float* __restrict__ stats,
                           const float* __restrict__ g, const float* __restrict__ b,
                           const unsigned short* __restrict__ res,
                           unsigned short* __restrict__ y,
                           int N, int C, int relu) {
    int i = blockIdx.x * blockDim.x + threadIdx.x;
    if (i >= N * C) return;
    int c = i % C;
    float inv = 1.0f / (float)N;
    float mean = stats[c] * inv;
    float var  = stats[128 + c] * inv - mean * mean;
    float rstd = rsqrtf(var + 1e-4f);
    float v = (x[i] - mean) * rstd * g[c] + b[c];
    if (res) v += bf2f(res[i]);
    if (relu) v = fmaxf(v, 0.0f);
    y[i] = f2bf(v);
}

// ---------------------------------------------------------------------------
// final segment-sum over batch (bf16 input) + scale by 1/vol
// ---------------------------------------------------------------------------
__global__ void k_seg(const unsigned short* __restrict__ x, const int* __restrict__ bidx,
                      float* __restrict__ osum, int N, int C) {
    int i = blockIdx.x * blockDim.x + threadIdx.x;
    if (i >= N * C) return;
    int n = i / C, c = i % C;
    atomicAdd(&osum[(size_t)bidx[n] * C + c], bf2f(x[i]));
}

__global__ void k_scale(const float* __restrict__ osum, float* __restrict__ out,
                        int n, float s) {
    int i = blockIdx.x * blockDim.x + threadIdx.x;
    if (i < n) out[i] = osum[i] * s;
}

// ---------------------------------------------------------------------------
// host launcher
// ---------------------------------------------------------------------------
// Input flattening assumption (setup_inputs insertion order; params flattened
// jax-style with dict keys sorted):
//   0: feats
//   1..9:   down l=0..2 : b, g, w
//   10..57: series l=0..3, block 0..1 : b1, b2, g1, g2, w1, w2
//   58: w_init   59: nbr_init   60..63: nbrs[0..3]   64..66: ds[0..2]
//   67: batch_idx   68: batch_size
extern "C" void kernel_launch(void* const* d_in, const int* in_sizes, int n_in,
                              void* d_out, int out_size, void* d_ws, size_t ws_size,
                              hipStream_t stream) {
    (void)n_in; (void)ws_size;
    const int CH[4] = {32, 64, 96, 128};

    const float* feats = (const float*)d_in[0];
    const float *down_b[3], *down_g[3], *down_w[3];
    for (int l = 0; l < 3; ++l) {
        down_b[l] = (const float*)d_in[1 + 3 * l];
        down_g[l] = (const float*)d_in[2 + 3 * l];
        down_w[l] = (const float*)d_in[3 + 3 * l];
    }
    const float *sb1[4][2], *sb2[4][2], *sg1[4][2], *sg2[4][2], *sw1[4][2], *sw2[4][2];
    for (int l = 0; l < 4; ++l)
        for (int b = 0; b < 2; ++b) {
            int base = 10 + (l * 2 + b) * 6;
            sb1[l][b] = (const float*)d_in[base + 0];
            sb2[l][b] = (const float*)d_in[base + 1];
            sg1[l][b] = (const float*)d_in[base + 2];
            sg2[l][b] = (const float*)d_in[base + 3];
            sw1[l][b] = (const float*)d_in[base + 4];
            sw2[l][b] = (const float*)d_in[base + 5];
        }
    const float* w_init   = (const float*)d_in[58];
    const int*   nbr_init = (const int*)d_in[59];
    const int*   nbrs[4];
    for (int l = 0; l < 4; ++l) nbrs[l] = (const int*)d_in[60 + l];
    const int* ds[3];
    for (int l = 0; l < 3; ++l) ds[l] = (const int*)d_in[64 + l];
    const int* bidx = (const int*)d_in[67];

    // recover per-level site counts from rulebook sizes
    int N[4];
    N[0] = in_sizes[59] / 125;
    N[1] = in_sizes[64] / 8;
    N[2] = in_sizes[65] / 8;
    N[3] = in_sizes[66] / 8;
    const int B = out_size / CH[3];            // 4

    // ---- carve workspace (offsets in floats; all region sizes 16B-aligned)
    float* arena = (float*)d_ws;
    size_t off = 0;
    unsigned short* xl[4];                     // bf16 activations per level
    size_t maxNC = 0;
    for (int l = 0; l < 4; ++l) {
        xl[l] = (unsigned short*)(arena + off);
        size_t nc = (size_t)N[l] * CH[l];      // multiple of 32
        off += nc / 2;                         // bf16 elements
        if (nc > maxNC) maxNC = nc;
    }
    float* tbuf = arena + off; off += maxNC;   // f32 conv output (pre-BN)
    unsigned short* ubuf = (unsigned short*)(arena + off);
    off += maxNC / 2;                          // bf16 intra-block activation
    unsigned short* wscr = (unsigned short*)(arena + off);
    off += (size_t)(27 * 128 * 128) / 2;       // bf16 weight scratch (max conv)
    float* stats = arena + off; off += 256;    // [sum(128) | sumsq(128)]
    float* osum  = arena + off; off += (size_t)B * 128;
    unsigned short* zrow = (unsigned short*)(arena + off);
    off += 64;                                 // 256B zero row for dead taps

    auto zero = [&](float* p, int n) {
        k_zero<<<dim3((n + 255) / 256), dim3(256), 0, stream>>>(p, n);
    };
    auto repack = [&](const float* w, int K, int Ci, int Co) {
        int tot = K * Ci * Co;
        k_repack_w<<<dim3((tot + 255) / 256), dim3(256), 0, stream>>>(w, wscr, K, Ci, Co);
    };
    // square res-block convs (Cin == Cout == CH[l], K = 27)
    auto conv_sq = [&](int l, const unsigned short* xin, const int* nb,
                       float* yo, int Nout) {
        dim3 g((Nout + 63) / 64), t(128);
        switch (l) {
        case 0:  k_conv_wmma< 32,  32, 27><<<g, t, 0, stream>>>(xin, nb, wscr, zrow, yo, Nout); break;
        case 1:  k_conv_wmma< 64,  64, 27><<<g, t, 0, stream>>>(xin, nb, wscr, zrow, yo, Nout); break;
        case 2:  k_conv_wmma< 96,  96, 27><<<g, t, 0, stream>>>(xin, nb, wscr, zrow, yo, Nout); break;
        default: k_conv_wmma<128, 128, 27><<<g, t, 0, stream>>>(xin, nb, wscr, zrow, yo, Nout); break;
        }
    };
    // strided downsample convs (CH[l] -> CH[l+1], K = 8)
    auto conv_dn = [&](int l, const unsigned short* xin, const int* nb,
                       float* yo, int Nout) {
        dim3 g((Nout + 63) / 64), t(128);
        switch (l) {
        case 0:  k_conv_wmma<32,  64, 8><<<g, t, 0, stream>>>(xin, nb, wscr, zrow, yo, Nout); break;
        case 1:  k_conv_wmma<64,  96, 8><<<g, t, 0, stream>>>(xin, nb, wscr, zrow, yo, Nout); break;
        default: k_conv_wmma<96, 128, 8><<<g, t, 0, stream>>>(xin, nb, wscr, zrow, yo, Nout); break;
        }
    };
    auto bnorm = [&](const float* x, const float* g, const float* b,
                     const unsigned short* res, unsigned short* y,
                     int n, int C, int relu) {
        zero(stats, 256);
        int total = n * C;
        int blocks = (total + 255) / 256;
        if (blocks > 2048) blocks = 2048;
        k_bn_stats<<<dim3(blocks), dim3(256), 0, stream>>>(x, stats, n, C);
        k_bn_apply<<<dim3((total + 255) / 256), dim3(256), 0, stream>>>(
            x, stats, g, b, res, y, n, C, relu);
    };

    // zero row used by async gather for inactive rulebook entries
    zero((float*)zrow, 32);

    // ---- initial 5^3 conv (Cin=1) ----
    k_conv_init<<<dim3((N[0] + 3) / 4), dim3(128), 0, stream>>>(
        feats, nbr_init, w_init, xl[0], N[0]);

    // ---- levels ----
    for (int l = 0; l < 4; ++l) {
        int C = CH[l], Nl = N[l];
        for (int b = 0; b < 2; ++b) {
            repack(sw1[l][b], 27, C, C);
            conv_sq(l, xl[l], nbrs[l], tbuf, Nl);
            bnorm(tbuf, sg1[l][b], sb1[l][b], (const unsigned short*)0, ubuf, Nl, C, 1);
            repack(sw2[l][b], 27, C, C);
            conv_sq(l, ubuf, nbrs[l], tbuf, Nl);
            bnorm(tbuf, sg2[l][b], sb2[l][b], xl[l], xl[l], Nl, C, 1);  // +res, relu
        }
        if (l < 3) {
            repack(down_w[l], 8, CH[l], CH[l + 1]);
            conv_dn(l, xl[l], ds[l], tbuf, N[l + 1]);
            bnorm(tbuf, down_g[l], down_b[l], (const unsigned short*)0, xl[l + 1],
                  N[l + 1], CH[l + 1], 1);
        }
    }

    // ---- segment mean over dense volume (192^3) ----
    zero(osum, B * 128);
    int tot = N[3] * 128;
    k_seg<<<dim3((tot + 255) / 256), dim3(256), 0, stream>>>(xl[3], bidx, osum, N[3], 128);
    k_scale<<<dim3((B * 128 + 255) / 256), dim3(256), 0, stream>>>(
        osum, (float*)d_out, B * 128, 1.0f / 7077888.0f);
}
